// GPRGNN_41729902247982
// MI455X (gfx1250) — matrix operand order repaired and checked by reading, not compiled, not run on hardware
//
#include <hip/hip_runtime.h>
#include <hip/hip_bf16.h>

#define NNODES 100000
#define NEDGES 1600000
#define NFEAT  512
#define NHID   256
#define NCLASS 32
#define KHOPS  10

typedef __bf16 bf16_t;
typedef __attribute__((ext_vector_type(16))) __bf16 v16bf;
typedef __attribute__((ext_vector_type(8)))  __bf16 v8bf;
typedef __attribute__((ext_vector_type(4)))  __bf16 v4bf;
typedef __attribute__((ext_vector_type(8)))  float  v8f;

#define BM  128
#define BK  32
#define BKP 48   // padded LDS row (96 B: 16B-aligned, spreads banks)

// ---------------- weight transpose + fp32 -> bf16 ----------------
__global__ void k_convert_transpose(const float* __restrict__ W, bf16_t* __restrict__ WT,
                                    int K, int Nc) {
    int idx = blockIdx.x * blockDim.x + threadIdx.x;
    if (idx < K * Nc) {
        int k = idx / Nc, n = idx - k * Nc;
        WT[n * K + k] = (bf16_t)W[idx];
    }
}

// ---------------- GEMM1: h = relu(x @ Ws + b), bf16 WMMA, fp32 accum ----------------
__global__ __launch_bounds__(256)
void k_gemm1(const float* __restrict__ X, const bf16_t* __restrict__ WT,
             const float* __restrict__ bias, bf16_t* __restrict__ H) {
    __shared__ __align__(16) bf16_t As[2][BM][BKP];
    __shared__ __align__(16) bf16_t Bs[2][128][BKP];

    const int row0 = blockIdx.x * BM;
    const int n0   = blockIdx.y * 128;
    const int tid  = threadIdx.x;
    const int wave = tid >> 5;
    const int lane = tid & 31;
    const int half = lane >> 4;
    const int lrow = lane & 15;
    const int mrow = wave * 16;

    // ---- k-invariant staging coordinates ----
    int aR[4], aC[4], aBase[4];
    float aM[4];
#pragma unroll
    for (int i = 0; i < 4; ++i) {
        int q  = i * 256 + tid;
        aR[i]  = q >> 3;
        aC[i]  = (q & 7) << 2;
        int gr = row0 + aR[i];
        int grc = gr < NNODES ? gr : (NNODES - 1);
        aM[i] = gr < NNODES ? 1.0f : 0.0f;
        aBase[i] = grc * NFEAT + aC[i];
    }
    int bN[2], bC[2], bBase[2];
#pragma unroll
    for (int i = 0; i < 2; ++i) {
        int o  = i * 256 + tid;
        bN[i]  = o >> 2;
        bC[i]  = (o & 3) << 3;
        bBase[i] = (n0 + bN[i]) * NFEAT + bC[i];
    }
    const int akb = half * 8;
    const int bkb = half * 8;

    v8f acc[8];
#pragma unroll
    for (int c = 0; c < 8; ++c)
#pragma unroll
        for (int r = 0; r < 8; ++r) acc[c][r] = 0.0f;

    float4 va[4];
    v8bf   vb[2];

    auto load_regs = [&](int k0) {
#pragma unroll
        for (int i = 0; i < 4; ++i) va[i] = *(const float4*)&X[aBase[i] + k0];
#pragma unroll
        for (int i = 0; i < 2; ++i) vb[i] = *(const v8bf*)&WT[bBase[i] + k0];
    };
    auto store_lds = [&](int buf) {
#pragma unroll
        for (int i = 0; i < 4; ++i) {
            v4bf p;
            p[0] = (bf16_t)(va[i].x * aM[i]);
            p[1] = (bf16_t)(va[i].y * aM[i]);
            p[2] = (bf16_t)(va[i].z * aM[i]);
            p[3] = (bf16_t)(va[i].w * aM[i]);
            *(v4bf*)&As[buf][aR[i]][aC[i]] = p;
        }
#pragma unroll
        for (int i = 0; i < 2; ++i) *(v8bf*)&Bs[buf][bN[i]][bC[i]] = vb[i];
    };
    auto compute = [&](int buf) {
        v8bf alo = *(const v8bf*)&As[buf][mrow + lrow][akb];
        v8bf ahi = *(const v8bf*)&As[buf][mrow + lrow][akb + 16];
        v16bf a = __builtin_shufflevector(alo, ahi, 0,1,2,3,4,5,6,7,8,9,10,11,12,13,14,15);
        v16bf b[8];
#pragma unroll
        for (int c = 0; c < 8; ++c) {
            v8bf blo = *(const v8bf*)&Bs[buf][c * 16 + lrow][bkb];
            v8bf bhi = *(const v8bf*)&Bs[buf][c * 16 + lrow][bkb + 16];
            b[c] = __builtin_shufflevector(blo, bhi, 0,1,2,3,4,5,6,7,8,9,10,11,12,13,14,15);
        }
#pragma unroll
        for (int c = 0; c < 8; ++c) {
            acc[c] = __builtin_amdgcn_wmma_f32_16x16x32_bf16(
                false, a, false, b[c], (short)0, acc[c], false, false);
        }
    };

    // ---- double-buffered main loop ----
    const int KT = NFEAT / BK;           // 16
    load_regs(0);
    store_lds(0);
    __syncthreads();
    for (int kt = 0; kt < KT - 1; ++kt) {
        load_regs((kt + 1) * BK);        // overlap with compute below
        compute(kt & 1);
        store_lds((kt + 1) & 1);
        __syncthreads();
    }
    compute((KT - 1) & 1);

    // ---- epilogue: uniform fast path for full blocks ----
    if (row0 + BM <= NNODES) {
#pragma unroll
        for (int c = 0; c < 8; ++c) {
            int gcol = n0 + c * 16 + lrow;
            float bv = bias[gcol];
#pragma unroll
            for (int r = 0; r < 8; ++r) {
                int grow = row0 + mrow + half * 8 + r;
                float v = acc[c][r] + bv;
                H[grow * NHID + gcol] = (bf16_t)(v > 0.0f ? v : 0.0f);
            }
        }
    } else {
#pragma unroll
        for (int c = 0; c < 8; ++c) {
            int gcol = n0 + c * 16 + lrow;
            float bv = bias[gcol];
#pragma unroll
            for (int r = 0; r < 8; ++r) {
                int grow = row0 + mrow + half * 8 + r;
                if (grow < NNODES) {
                    float v = acc[c][r] + bv;
                    H[grow * NHID + gcol] = (bf16_t)(v > 0.0f ? v : 0.0f);
                }
            }
        }
    }
}

// ---------------- GEMM2: h2 = relu(h @ We + b), fp32 out ----------------
__global__ __launch_bounds__(256)
void k_gemm2(const bf16_t* __restrict__ H, const bf16_t* __restrict__ WT,
             const float* __restrict__ bias, float* __restrict__ H2) {
    __shared__ __align__(16) bf16_t As[2][BM][BKP];
    __shared__ __align__(16) bf16_t Bs[2][NCLASS][BKP];

    const int row0 = blockIdx.x * BM;
    const int tid  = threadIdx.x;
    const int wave = tid >> 5;
    const int lane = tid & 31;
    const int half = lane >> 4;
    const int lrow = lane & 15;
    const int mrow = wave * 16;

    const v8bf zero8 = (v8bf)(__bf16)0.0f;

    int aR[2], aC[2], aBase[2];
    bool aOk[2];
#pragma unroll
    for (int i = 0; i < 2; ++i) {
        int o  = i * 256 + tid;
        aR[i]  = o >> 2;
        aC[i]  = (o & 3) << 3;
        int gr = row0 + aR[i];
        int grc = gr < NNODES ? gr : (NNODES - 1);
        aOk[i] = gr < NNODES;
        aBase[i] = grc * NHID + aC[i];
    }
    const bool bAct = tid < 128;
    const int bN = tid >> 2;
    const int bC = (tid & 3) << 3;
    const int bBase = bN * NHID + bC;
    const int akb = half * 8;
    const int bkb = half * 8;

    v8f acc[2];
#pragma unroll
    for (int c = 0; c < 2; ++c)
#pragma unroll
        for (int r = 0; r < 8; ++r) acc[c][r] = 0.0f;

    v8bf va[2], vb;

    auto load_regs = [&](int k0) {
#pragma unroll
        for (int i = 0; i < 2; ++i) {
            v8bf v = *(const v8bf*)&H[aBase[i] + k0];
            va[i] = aOk[i] ? v : zero8;
        }
        vb = bAct ? *(const v8bf*)&WT[bBase + k0] : zero8;
    };
    auto store_lds = [&](int buf) {
#pragma unroll
        for (int i = 0; i < 2; ++i) *(v8bf*)&As[buf][aR[i]][aC[i]] = va[i];
        if (bAct) *(v8bf*)&Bs[buf][bN][bC] = vb;
    };
    auto compute = [&](int buf) {
        v8bf alo = *(const v8bf*)&As[buf][mrow + lrow][akb];
        v8bf ahi = *(const v8bf*)&As[buf][mrow + lrow][akb + 16];
        v16bf a = __builtin_shufflevector(alo, ahi, 0,1,2,3,4,5,6,7,8,9,10,11,12,13,14,15);
        v16bf b[2];
#pragma unroll
        for (int c = 0; c < 2; ++c) {
            v8bf blo = *(const v8bf*)&Bs[buf][c * 16 + lrow][bkb];
            v8bf bhi = *(const v8bf*)&Bs[buf][c * 16 + lrow][bkb + 16];
            b[c] = __builtin_shufflevector(blo, bhi, 0,1,2,3,4,5,6,7,8,9,10,11,12,13,14,15);
        }
#pragma unroll
        for (int c = 0; c < 2; ++c) {
            acc[c] = __builtin_amdgcn_wmma_f32_16x16x32_bf16(
                false, a, false, b[c], (short)0, acc[c], false, false);
        }
    };

    const int KT = NHID / BK;            // 8
    load_regs(0);
    store_lds(0);
    __syncthreads();
    for (int kt = 0; kt < KT - 1; ++kt) {
        load_regs((kt + 1) * BK);
        compute(kt & 1);
        store_lds((kt + 1) & 1);
        __syncthreads();
    }
    compute((KT - 1) & 1);

    if (row0 + BM <= NNODES) {
#pragma unroll
        for (int c = 0; c < 2; ++c) {
            int gcol = c * 16 + lrow;
            float bv = bias[gcol];
#pragma unroll
            for (int r = 0; r < 8; ++r) {
                int grow = row0 + mrow + half * 8 + r;
                float v = acc[c][r] + bv;
                H2[grow * NCLASS + gcol] = v > 0.0f ? v : 0.0f;
            }
        }
    } else {
#pragma unroll
        for (int c = 0; c < 2; ++c) {
            int gcol = c * 16 + lrow;
            float bv = bias[gcol];
#pragma unroll
            for (int r = 0; r < 8; ++r) {
                int grow = row0 + mrow + half * 8 + r;
                if (grow < NNODES) {
                    float v = acc[c][r] + bv;
                    H2[grow * NCLASS + gcol] = v > 0.0f ? v : 0.0f;
                }
            }
        }
    }
}

// ---------------- graph normalization ----------------
__global__ void k_deg_init(float* deg, int n) {          // self-loop weight 1
    int i = blockIdx.x * blockDim.x + threadIdx.x;
    if (i < n) deg[i] = 1.0f;
}
__global__ void k_deg_edges(const int* __restrict__ ei, const float* __restrict__ ew,
                            float* deg, int e) {
    int i = blockIdx.x * blockDim.x + threadIdx.x;
    if (i < e) atomicAdd(&deg[ei[NEDGES + i]], ew[i]);   // col = ei[1][i]
}
__global__ void k_deg_to_dis(float* deg, int n) {
    int i = blockIdx.x * blockDim.x + threadIdx.x;
    if (i < n) { float d = deg[i]; deg[i] = d > 0.0f ? rsqrtf(d) : 0.0f; }
}
__global__ void k_norm(const int* __restrict__ ei, const float* __restrict__ ew,
                       const float* __restrict__ dis, float* __restrict__ nrm, int e) {
    int i = blockIdx.x * blockDim.x + threadIdx.x;
    if (i < e) nrm[i] = dis[ei[i]] * ew[i] * dis[ei[NEDGES + i]];
}

// ---------------- propagation ----------------
__global__ void k_hidden_init(const float* __restrict__ h2, const float* __restrict__ temp,
                              float* __restrict__ out, int n) {
    int i = blockIdx.x * blockDim.x + threadIdx.x;
    if (i < n) out[i] = temp[0] * h2[i];
}
__global__ void k_spmm_self(const float* __restrict__ xk, const float* __restrict__ dis,
                            float* __restrict__ xnew, int n) {
    int i = blockIdx.x * blockDim.x + threadIdx.x;
    if (i < n) { float d = dis[i >> 5]; xnew[i] = d * d * xk[i]; }
}
__global__ void k_spmm_edges(const int* __restrict__ ei, const float* __restrict__ nrm,
                             const float* __restrict__ xk, float* __restrict__ xnew, int e) {
    int g = blockIdx.x * blockDim.x + threadIdx.x;
    int i = g >> 5;            // edge (one wave per edge)
    int f = g & 31;            // feature lane
    if (i < e) {
        int r = ei[i], c = ei[NEDGES + i];
        atomicAdd(&xnew[c * NCLASS + f], nrm[i] * xk[r * NCLASS + f]);
    }
}
__global__ void k_accum(const float* __restrict__ xnew, const float* __restrict__ temp,
                        int kidx, float* __restrict__ out, int n) {
    int i = blockIdx.x * blockDim.x + threadIdx.x;
    if (i < n) out[i] += temp[kidx] * xnew[i];
}

// ---------------- launcher ----------------
extern "C" void kernel_launch(void* const* d_in, const int* in_sizes, int n_in,
                              void* d_out, int out_size, void* d_ws, size_t ws_size,
                              hipStream_t stream) {
    const float* x         = (const float*)d_in[0];
    const float* edge_attr = (const float*)d_in[1];
    const float* W_start   = (const float*)d_in[2];
    const float* b_start   = (const float*)d_in[3];
    const float* W_end     = (const float*)d_in[4];
    const float* b_end     = (const float*)d_in[5];
    const float* temp      = (const float*)d_in[6];
    const int*   eidx      = (const int*)d_in[7];
    float* out = (float*)d_out;

    char* ws = (char*)d_ws;
    bf16_t* WTs = (bf16_t*)(ws);                         // 512*256 bf16  = 262144 B
    bf16_t* WTe = (bf16_t*)(ws + 262144);                // 256*32 bf16   = 16384 B
    bf16_t* Hb  = (bf16_t*)(ws + 278528);                // N*256 bf16    = 51200000 B
    float*  H2  = (float*)(ws + 51478528);               // N*32 f32      = 12800000 B
    float*  dis = (float*)(ws + 64278528);               // N f32 (deg->dis in place)
    float*  nrm = (float*)(ws + 64678528);               // E f32         = 6400000 B
    float*  xkA = (float*)(ws + 71078528);               // N*32 f32
    float*  xkB = (float*)(ws + 83878528);               // N*32 f32  (end 96678528 B)

    // MLP front-end
    k_convert_transpose<<<(NFEAT * NHID + 255) / 256, 256, 0, stream>>>(W_start, WTs, NFEAT, NHID);
    k_convert_transpose<<<(NHID * NCLASS + 255) / 256, 256, 0, stream>>>(W_end, WTe, NHID, NCLASS);
    dim3 g1((NNODES + BM - 1) / BM, NHID / 128);
    k_gemm1<<<g1, 256, 0, stream>>>(x, WTs, b_start, Hb);
    k_gemm2<<<(NNODES + BM - 1) / BM, 256, 0, stream>>>(Hb, WTe, b_end, H2);

    // gcn_norm
    k_deg_init<<<(NNODES + 255) / 256, 256, 0, stream>>>(dis, NNODES);
    k_deg_edges<<<(NEDGES + 255) / 256, 256, 0, stream>>>(eidx, edge_attr, dis, NEDGES);
    k_deg_to_dis<<<(NNODES + 255) / 256, 256, 0, stream>>>(dis, NNODES);
    k_norm<<<(NEDGES + 255) / 256, 256, 0, stream>>>(eidx, edge_attr, dis, nrm, NEDGES);

    // GPR propagation
    const int nelem = NNODES * NCLASS;
    k_hidden_init<<<(nelem + 255) / 256, 256, 0, stream>>>(H2, temp, out, nelem);
    const float* xk = H2;
    float* bufs[2] = {xkA, xkB};
    for (int k = 0; k < KHOPS; ++k) {
        float* xnew = bufs[k & 1];
        k_spmm_self<<<(nelem + 255) / 256, 256, 0, stream>>>(xk, dis, xnew, nelem);
        long eg = (long)NEDGES * 32;
        k_spmm_edges<<<(int)((eg + 255) / 256), 256, 0, stream>>>(eidx, nrm, xk, xnew, NEDGES);
        k_accum<<<(nelem + 255) / 256, 256, 0, stream>>>(xnew, temp, k + 1, out, nelem);
        xk = xnew;
    }
}
